// MNIST_paper_88579405513301
// MI455X (gfx1250) — matrix operand order repaired and checked by reading, not compiled, not run on hardware
//
#include <hip/hip_runtime.h>
#include <hip/hip_bf16.h>
#include <math.h>

typedef __hip_bfloat16 bf16_t;
typedef __attribute__((ext_vector_type(16))) __bf16 v16bf;
typedef __attribute__((ext_vector_type(8)))  float v8f;
typedef __attribute__((ext_vector_type(4)))  unsigned int v4u;

union FragU { v16bf v; v4u u[2]; };

// ---------------------------------------------------------------------------
// Fragment loaders (bf16, 16x16x32 WMMA).
// A (16xK): lane r(0..15)/half h: elems 0-7 = K[k+8h .. k+8h+7], elems 8-15 =
//           K[k+16+8h .. +7]  (ISA 16-bit A 16x32 table).
// B (Kx16): lane n holds column n; half h: elems 0-7 = K[k+8h .. k+8h+7],
//           elems 8-15 = K[k+16+8h .. +7] (same order as A).
// Both operands are rows of a row-major [rows x stride] bf16 matrix.
// ---------------------------------------------------------------------------
__device__ __forceinline__ v16bf load_fragA(const bf16_t* __restrict__ base, int stride, int k) {
  const int lane = threadIdx.x & 31;
  const int r = lane & 15;
  const int h = lane >> 4;
  const bf16_t* p = base + (size_t)r * stride + k + h * 8;
  FragU f;
  f.u[0] = *(const v4u*)(p);
  f.u[1] = *(const v4u*)(p + 16);
  return f.v;
}

__device__ __forceinline__ v16bf load_fragB(const bf16_t* __restrict__ base, int stride, int k) {
  const int lane = threadIdx.x & 31;
  const int n = lane & 15;
  const int h = lane >> 4;
  const bf16_t* p = base + (size_t)n * stride + k + h * 8;
  FragU f;
  f.u[0] = *(const v4u*)(p);
  f.u[1] = *(const v4u*)(p + 16);
  return f.v;
}

__device__ __forceinline__ v8f wmma_bf16(v16bf a, v16bf b, v8f c) {
  // (neg_a, A, neg_b, B, c_mod, C, reuse_a, reuse_b)
  return __builtin_amdgcn_wmma_f32_16x16x32_bf16(false, a, false, b, (short)0, c, false, false);
}

// ACT: 0=none, 1=relu, 2=leaky(0.1), 3=tanh
template<int ACT>
__device__ __forceinline__ float apply_act(float x) {
  if (ACT == 1) return x > 0.0f ? x : 0.0f;
  if (ACT == 2) return x > 0.0f ? x : 0.1f * x;
  if (ACT == 3) return tanhf(x);
  return x;
}

// C 16x16 f32 layout: VGPR i, lanes 0-15 -> row i, lanes 16-31 -> row 8+i;
// column = lane&15.
template<int ACT>
__device__ __forceinline__ void store_tile(
    v8f c, int m0, int n0, const float* __restrict__ bias,
    float* __restrict__ outF, int ldoF,
    float* __restrict__ outF2, int ldoF2,
    bf16_t* __restrict__ outB, int ldoB, int N)
{
  const int lane = threadIdx.x & 31;
  const int col = lane & 15;
  const int h = lane >> 4;
  const int n = n0 + col;
  const bool valid = n < N;
  const float bv = valid ? bias[n] : 0.0f;
#pragma unroll
  for (int i = 0; i < 8; ++i) {
    const int m = m0 + h * 8 + i;
    const float v = apply_act<ACT>(c[i] + bv);
    if (outF  && valid) outF [(size_t)m * ldoF  + n] = v;
    if (outF2 && valid) outF2[(size_t)m * ldoF2 + n] = v;
    if (outB && n < ldoB) outB[(size_t)m * ldoB + n] = __float2bfloat16(valid ? v : 0.0f);
  }
}

// out[m,n] = act( sum_k A[m,k]*W[n,k] + bias[n] )
// Each wave: 32x64 tile (2x4 WMMA tiles, 8 v_wmma per K-step from 6 fragment
// loads -> 0.75 VMEM/WMMA). Block: 8 waves stacked in M = 256 rows x 64 cols.
// Grid: (ceil(N/64), M/256). W is padded to [gridX*64, Kpad] bf16 with zeros.
template<int ACT>
__global__ __launch_bounds__(256) void gemm_bf16_kernel(
    const bf16_t* __restrict__ A, int lda,
    const bf16_t* __restrict__ W, int ldw,
    const float* __restrict__ bias,
    float* __restrict__ outF, int ldoF,
    float* __restrict__ outF2, int ldoF2,
    bf16_t* __restrict__ outB, int ldoB,
    int N, int Kpad)
{
  const int waveId = threadIdx.x >> 5;
  const int m0 = blockIdx.y * 256 + waveId * 32;
  const int n0 = blockIdx.x * 64;

  const bf16_t* A0 = A + (size_t)m0 * lda;
  const bf16_t* A1 = A + (size_t)(m0 + 16) * lda;
  const bf16_t* W0 = W + (size_t)n0 * ldw;
  const bf16_t* W1 = W + (size_t)(n0 + 16) * ldw;
  const bf16_t* W2 = W + (size_t)(n0 + 32) * ldw;
  const bf16_t* W3 = W + (size_t)(n0 + 48) * ldw;

  v8f c00 = {}, c01 = {}, c02 = {}, c03 = {};
  v8f c10 = {}, c11 = {}, c12 = {}, c13 = {};
  for (int k = 0; k < Kpad; k += 32) {
    v16bf a0 = load_fragA(A0, lda, k);
    v16bf a1 = load_fragA(A1, lda, k);
    v16bf b0 = load_fragB(W0, ldw, k);
    v16bf b1 = load_fragB(W1, ldw, k);
    v16bf b2 = load_fragB(W2, ldw, k);
    v16bf b3 = load_fragB(W3, ldw, k);
    c00 = wmma_bf16(a0, b0, c00);
    c01 = wmma_bf16(a0, b1, c01);
    c02 = wmma_bf16(a0, b2, c02);
    c03 = wmma_bf16(a0, b3, c03);
    c10 = wmma_bf16(a1, b0, c10);
    c11 = wmma_bf16(a1, b1, c11);
    c12 = wmma_bf16(a1, b2, c12);
    c13 = wmma_bf16(a1, b3, c13);
  }
  store_tile<ACT>(c00, m0,      n0,      bias, outF, ldoF, outF2, ldoF2, outB, ldoB, N);
  store_tile<ACT>(c01, m0,      n0 + 16, bias, outF, ldoF, outF2, ldoF2, outB, ldoB, N);
  store_tile<ACT>(c02, m0,      n0 + 32, bias, outF, ldoF, outF2, ldoF2, outB, ldoB, N);
  store_tile<ACT>(c03, m0,      n0 + 48, bias, outF, ldoF, outF2, ldoF2, outB, ldoB, N);
  store_tile<ACT>(c10, m0 + 16, n0,      bias, outF, ldoF, outF2, ldoF2, outB, ldoB, N);
  store_tile<ACT>(c11, m0 + 16, n0 + 16, bias, outF, ldoF, outF2, ldoF2, outB, ldoB, N);
  store_tile<ACT>(c12, m0 + 16, n0 + 32, bias, outF, ldoF, outF2, ldoF2, outB, ldoB, N);
  store_tile<ACT>(c13, m0 + 16, n0 + 48, bias, outF, ldoF, outF2, ldoF2, outB, ldoB, N);
}

// For each query row q: idxOut[q] = argmin_t ( t2[t] - 2 * dot(Q[q], T[t]) )
// Z = 64. A operand = target tile (rows), B operand = query block (cols), so
// each lane's accumulator holds 8 target rows for one query column -> the
// argmin reduction is register-resident; one shfl_xor(16) merges halves.
__global__ __launch_bounds__(256) void nearest_kernel(
    const bf16_t* __restrict__ Qd, const bf16_t* __restrict__ Td,
    const float* __restrict__ t2, int* __restrict__ idxOut, int Q, int T)
{
  const int wave = blockIdx.x * (blockDim.x >> 5) + (threadIdx.x >> 5);
  const int q0 = wave * 16;
  if (q0 >= Q) return;                   // wave-uniform: EXEC stays all-ones
  const int lane = threadIdx.x & 31;
  const int h = lane >> 4;

  const bf16_t* qbase = Qd + (size_t)q0 * 64;
  v16bf qb0 = load_fragB(qbase, 64, 0);
  v16bf qb1 = load_fragB(qbase, 64, 32);

  float best = 3.402823466e38f;
  int bestIdx = 0;
  for (int t0 = 0; t0 < T; t0 += 16) {
    const bf16_t* tbase = Td + (size_t)t0 * 64;
    v16bf ta0 = load_fragA(tbase, 64, 0);
    v16bf ta1 = load_fragA(tbase, 64, 32);
    v8f c = {};
    c = wmma_bf16(ta0, qb0, c);
    c = wmma_bf16(ta1, qb1, c);
    const int rbase = t0 + h * 8;
#pragma unroll
    for (int i = 0; i < 8; ++i) {
      const float val = t2[rbase + i] - 2.0f * c[i];
      if (val < best) { best = val; bestIdx = rbase + i; }  // strict < keeps first min
    }
  }
  const float ov = __shfl_xor(best, 16, 32);
  const int   oi = __shfl_xor(bestIdx, 16, 32);
  if (ov < best || (ov == best && oi < bestIdx)) { best = ov; bestIdx = oi; }
  if (h == 0) idxOut[q0 + (lane & 15)] = bestIdx;
}

// ------------------------- small utility kernels ---------------------------
__global__ void pad_to_bf16(const float* __restrict__ src, bf16_t* __restrict__ dst,
                            int rows, int cols, int drows, int dcols) {
  const int i = blockIdx.x * blockDim.x + threadIdx.x;
  const int total = drows * dcols;
  if (i >= total) return;
  const int r = i / dcols, c = i - r * dcols;
  const float v = (r < rows && c < cols) ? src[(size_t)r * cols + c] : 0.0f;
  dst[i] = __float2bfloat16(v);
}

__global__ void row_sqnorm64(const bf16_t* __restrict__ x, float* __restrict__ out, int rows) {
  const int r = blockIdx.x * blockDim.x + threadIdx.x;
  if (r >= rows) return;
  const bf16_t* p = x + (size_t)r * 64;
  float s = 0.0f;
#pragma unroll 8
  for (int i = 0; i < 64; ++i) { const float v = __bfloat162float(p[i]); s += v * v; }
  out[r] = s;
}

__global__ void gather64_f32_to_bf16(const float* __restrict__ src, const int* __restrict__ idx,
                                     bf16_t* __restrict__ dst, int n) {
  const int i = blockIdx.x * blockDim.x + threadIdx.x;
  if (i >= n) return;
  const int m = i >> 6, c = i & 63;
  dst[i] = __float2bfloat16(src[(size_t)idx[m] * 64 + c]);
}

__global__ void gather64_f32(const float* __restrict__ src, const int* __restrict__ idx,
                             float* __restrict__ dst, int n) {
  const int i = blockIdx.x * blockDim.x + threadIdx.x;
  if (i >= n) return;
  const int m = i >> 6, c = i & 63;
  dst[i] = src[(size_t)idx[m] * 64 + c];
}

// ---------------------------------------------------------------------------
extern "C" void kernel_launch(void* const* d_in, const int* in_sizes, int n_in,
                              void* d_out, int out_size, void* d_ws, size_t ws_size,
                              hipStream_t stream) {
  (void)in_sizes; (void)n_in; (void)out_size; (void)ws_size;
  const float* X   = (const float*)d_in[0];
  const float* We1 = (const float*)d_in[1];  const float* be1 = (const float*)d_in[2];
  const float* We2 = (const float*)d_in[3];  const float* be2 = (const float*)d_in[4];
  const float* We3 = (const float*)d_in[5];  const float* be3 = (const float*)d_in[6];
  const float* We4 = (const float*)d_in[7];  const float* be4 = (const float*)d_in[8];
  const float* emb = (const float*)d_in[9];
  const float* Wd1 = (const float*)d_in[10]; const float* bd1 = (const float*)d_in[11];
  const float* Wd2 = (const float*)d_in[12]; const float* bd2 = (const float*)d_in[13];
  const float* Wd3 = (const float*)d_in[14]; const float* bd3 = (const float*)d_in[15];
  const float* Wd4 = (const float*)d_in[16]; const float* bd4 = (const float*)d_in[17];

  const size_t Bm = 16384, Kc = 4096;

  // output regions: (Z_dec, Z_enc, Z_dec, Z_enc_for_emb)
  float* out0 = (float*)d_out;                 // [B,784]
  float* out1 = out0 + Bm * 784;               // [B,64]
  float* out2 = out1 + Bm * 64;                // [B,784]
  float* out3 = out2 + Bm * 784;               // [K,64]

  // workspace bump allocator (256B aligned)
  char* ws = (char*)d_ws;
  size_t off = 0;
  auto wsalloc = [&](size_t bytes) -> void* {
    void* p = ws + off;
    off += (bytes + 255) & ~(size_t)255;
    return p;
  };

  bf16_t* Xp   = (bf16_t*)wsalloc(Bm * 800 * 2);
  bf16_t* h1b  = (bf16_t*)wsalloc(Bm * 1024 * 2);   // reused as d3
  bf16_t* h2b  = (bf16_t*)wsalloc(Bm * 512 * 2);    // reused as d2
  bf16_t* h3b  = (bf16_t*)wsalloc(Bm * 320 * 2);    // reused as d1
  bf16_t* zenb = (bf16_t*)wsalloc(Bm * 64 * 2);
  bf16_t* zemb = (bf16_t*)wsalloc(Bm * 64 * 2);
  bf16_t* embb = (bf16_t*)wsalloc(Kc * 64 * 2);
  bf16_t* We1p = (bf16_t*)wsalloc((size_t)1024 * 800 * 2);
  bf16_t* We2p = (bf16_t*)wsalloc((size_t)512 * 1024 * 2);
  bf16_t* We3p = (bf16_t*)wsalloc((size_t)320 * 512 * 2);
  bf16_t* We4p = (bf16_t*)wsalloc((size_t)64 * 320 * 2);
  bf16_t* Wd1p = (bf16_t*)wsalloc((size_t)320 * 64 * 2);
  bf16_t* Wd2p = (bf16_t*)wsalloc((size_t)512 * 320 * 2);
  bf16_t* Wd3p = (bf16_t*)wsalloc((size_t)1024 * 512 * 2);
  bf16_t* Wd4p = (bf16_t*)wsalloc((size_t)832 * 1024 * 2);
  float*  embsq = (float*)wsalloc(Kc * 4);
  float*  zsq   = (float*)wsalloc(Bm * 4);
  int*    idxQ  = (int*)wsalloc(Bm * 4);
  int*    idxT  = (int*)wsalloc(Kc * 4);

  auto pad = [&](const float* s, bf16_t* d, int r, int c, int dr, int dc) {
    const int total = dr * dc;
    pad_to_bf16<<<(total + 255) / 256, 256, 0, stream>>>(s, d, r, c, dr, dc);
  };

  // bf16 conversions (+ zero padding to K%32==0, N%64==0)
  pad(X,   Xp,   16384, 784, 16384, 800);
  pad(We1, We1p, 1000, 784, 1024, 800);
  pad(We2, We2p, 500, 1000, 512, 1024);
  pad(We3, We3p, 300, 500, 320, 512);
  pad(We4, We4p, 64, 300, 64, 320);
  pad(emb, embb, 4096, 64, 4096, 64);
  pad(Wd1, Wd1p, 300, 64, 320, 64);
  pad(Wd2, Wd2p, 500, 300, 512, 320);
  pad(Wd3, Wd3p, 1000, 500, 1024, 512);
  pad(Wd4, Wd4p, 784, 1000, 832, 1024);

  const dim3 blk(256);
  const int gy = 16384 / 256;

  // encoder
  gemm_bf16_kernel<1><<<dim3(16, gy), blk, 0, stream>>>(Xp, 800, We1p, 800, be1,
      nullptr, 0, nullptr, 0, h1b, 1024, 1000, 800);
  gemm_bf16_kernel<1><<<dim3(8, gy), blk, 0, stream>>>(h1b, 1024, We2p, 1024, be2,
      nullptr, 0, nullptr, 0, h2b, 512, 500, 1024);
  gemm_bf16_kernel<1><<<dim3(5, gy), blk, 0, stream>>>(h2b, 512, We3p, 512, be3,
      nullptr, 0, nullptr, 0, h3b, 320, 300, 512);
  gemm_bf16_kernel<0><<<dim3(1, gy), blk, 0, stream>>>(h3b, 320, We4p, 320, be4,
      out1, 64, nullptr, 0, zenb, 64, 64, 320);

  // vector quantization (both directions)
  row_sqnorm64<<<(4096 + 255) / 256, 256, 0, stream>>>(embb, embsq, 4096);
  row_sqnorm64<<<(16384 + 255) / 256, 256, 0, stream>>>(zenb, zsq, 16384);
  nearest_kernel<<<16384 / 128, 256, 0, stream>>>(zenb, embb, embsq, idxQ, 16384, 4096);
  nearest_kernel<<<4096 / 128, 256, 0, stream>>>(embb, zenb, zsq, idxT, 4096, 16384);
  gather64_f32_to_bf16<<<(16384 * 64 + 255) / 256, 256, 0, stream>>>(emb, idxQ, zemb, 16384 * 64);
  gather64_f32<<<(4096 * 64 + 255) / 256, 256, 0, stream>>>(out1, idxT, out3, 4096 * 64);

  // decoder (d1/d2/d3 reuse h3b/h2b/h1b)
  gemm_bf16_kernel<2><<<dim3(5, gy), blk, 0, stream>>>(zemb, 64, Wd1p, 64, bd1,
      nullptr, 0, nullptr, 0, h3b, 320, 300, 64);
  gemm_bf16_kernel<2><<<dim3(8, gy), blk, 0, stream>>>(h3b, 320, Wd2p, 320, bd2,
      nullptr, 0, nullptr, 0, h2b, 512, 500, 320);
  gemm_bf16_kernel<2><<<dim3(16, gy), blk, 0, stream>>>(h2b, 512, Wd3p, 512, bd3,
      nullptr, 0, nullptr, 0, h1b, 1024, 1000, 512);
  gemm_bf16_kernel<3><<<dim3(13, gy), blk, 0, stream>>>(h1b, 1024, Wd4p, 1024, bd4,
      out0, 784, out2, 784, nullptr, 0, 784, 1024);
}
